// AttnBlock_4475355922443
// MI455X (gfx1250) — matrix operand order repaired and checked by reading, not compiled, not run on hardware
//
#include <hip/hip_runtime.h>

// ---------------------------------------------------------------------------
// AttnBlock (VAE-style) for MI455X / gfx1250, wave32 + WMMA f16 pipeline.
//   GroupNorm -> q,k,v (1x1 conv GEMMs) -> S = Q^T K * scale -> softmax ->
//   O = V P^T -> proj GEMM + bias + residual.
//
// All intermediate layouts are chosen so that BOTH GEMM operands have unit
// K-stride => A/B fragments load as global_load_b128 (2 loads for A, 2 per
// B subtile).  Each wave computes a 16x64 D stripe (4 WMMAs per K step,
// A fragment reused 4x).  Softmax stages its 16KB row into LDS with
// gfx1250 global_load_async_to_lds_b128 (ASYNCcnt) before its 3 passes.
// ---------------------------------------------------------------------------

typedef __attribute__((ext_vector_type(16))) _Float16 v16h;
typedef __attribute__((ext_vector_type(8)))  _Float16 v8h;
typedef __attribute__((ext_vector_type(8)))  float    v8f;

#define BATCH 2
#define CCH   512
#define HW    4096
#define NGRP  8
#define CPG   (CCH / NGRP)   // 64
#define NSUB  4              // N subtiles per wave (16x64 D stripe)

// ---------------------------------------------------------------------------
// f32 -> f16 cast (weights)
// ---------------------------------------------------------------------------
__global__ void cast_f32_f16(const float* __restrict__ src,
                             _Float16* __restrict__ dst, int n) {
  int i = blockIdx.x * blockDim.x + threadIdx.x;
  if (i < n) dst[i] = (_Float16)src[i];
}

// ---------------------------------------------------------------------------
// GroupNorm pass 1: per (b,g) mean / rstd.  One block per (b,g); the group's
// data is a contiguous 64*4096-float chunk in [B,C,HW] layout.
// ---------------------------------------------------------------------------
__global__ void gn_stats(const float* __restrict__ x,
                         float* __restrict__ mean,
                         float* __restrict__ rstd) {
  const int bg = blockIdx.x;                 // 0..15
  const size_t n = (size_t)CPG * HW;
  const float* p = x + (size_t)bg * n;
  float s = 0.f, s2 = 0.f;
  for (size_t i = threadIdx.x; i < n; i += blockDim.x) {
    float v = p[i];
    s += v; s2 += v * v;
  }
  __shared__ float rs[256], rq[256];
  rs[threadIdx.x] = s; rq[threadIdx.x] = s2;
  __syncthreads();
  for (int off = 128; off > 0; off >>= 1) {
    if ((int)threadIdx.x < off) {
      rs[threadIdx.x] += rs[threadIdx.x + off];
      rq[threadIdx.x] += rq[threadIdx.x + off];
    }
    __syncthreads();
  }
  if (threadIdx.x == 0) {
    float m   = rs[0] / (float)n;
    float var = rq[0] / (float)n - m * m;
    mean[bg] = m;
    rstd[bg] = rsqrtf(var + 1e-6f);
  }
}

// ---------------------------------------------------------------------------
// GroupNorm pass 2: normalize + affine, emit f16 activations TRANSPOSED to
// token-major [B, HW, C] so downstream GEMMs see unit K-stride.
// ---------------------------------------------------------------------------
__global__ void gn_apply_t(const float* __restrict__ x,
                           const float* __restrict__ mean,
                           const float* __restrict__ rstd,
                           const float* __restrict__ gw,
                           const float* __restrict__ gb,
                           _Float16* __restrict__ xnT) {
  size_t i = (size_t)blockIdx.x * blockDim.x + threadIdx.x;   // over [B,HW,C]
  if (i >= (size_t)BATCH * CCH * HW) return;
  int c = (int)(i % CCH);
  int t = (int)((i / CCH) % HW);
  int b = (int)(i / ((size_t)CCH * HW));
  int bg = b * NGRP + c / CPG;
  float v = (x[((size_t)b * CCH + c) * HW + t] - mean[bg]) * rstd[bg] * gw[c] + gb[c];
  xnT[i] = (_Float16)v;
}

// ---------------------------------------------------------------------------
// Row softmax over 4096 columns; f32 in, f16 probabilities out (row-major).
// One block (256 threads) per row.  The 16KB row is staged into LDS with
// gfx1250 async loads (global_load_async_to_lds_b128 + s_wait_asynccnt),
// then the max / sum / write passes read LDS instead of global 3x.
// ---------------------------------------------------------------------------
__global__ void softmax_rows(const float* __restrict__ S,
                             _Float16* __restrict__ P) {
  __shared__ float row[HW];          // 16 KB
  __shared__ float red[256];
  const size_t r0 = (size_t)blockIdx.x * HW;
  const float* s = S + r0;
  _Float16*    p = P + r0;

  // --- async stage: 4 chunks x (256 threads x 16B) = 16 KB -----------------
  {
    unsigned ldsBase = (unsigned)(size_t)(&row[0]);   // low 32b = LDS offset
    unsigned tb = threadIdx.x * 16u;
#pragma unroll
    for (int c = 0; c < 4; ++c) {
      unsigned           loff = ldsBase + (unsigned)(c * 4096) + tb;
      unsigned long long ga   = (unsigned long long)(size_t)s
                              + (unsigned long long)(c * 4096) + tb;
      asm volatile("global_load_async_to_lds_b128 %0, %1, off"
                   :: "v"(loff), "v"(ga) : "memory");
    }
    asm volatile("s_wait_asynccnt 0x0" ::: "memory");
  }
  __syncthreads();

  float mx = -3.402823e38f;
  for (int j = threadIdx.x; j < HW; j += 256) mx = fmaxf(mx, row[j]);
  red[threadIdx.x] = mx; __syncthreads();
  for (int off = 128; off > 0; off >>= 1) {
    if ((int)threadIdx.x < off)
      red[threadIdx.x] = fmaxf(red[threadIdx.x], red[threadIdx.x + off]);
    __syncthreads();
  }
  mx = red[0]; __syncthreads();

  float sum = 0.f;
  for (int j = threadIdx.x; j < HW; j += 256) sum += __expf(row[j] - mx);
  red[threadIdx.x] = sum; __syncthreads();
  for (int off = 128; off > 0; off >>= 1) {
    if ((int)threadIdx.x < off) red[threadIdx.x] += red[threadIdx.x + off];
    __syncthreads();
  }
  float inv = 1.0f / red[0];

  for (int j = threadIdx.x; j < HW; j += 256)
    p[j] = (_Float16)(__expf(row[j] - mx) * inv);
}

// ---------------------------------------------------------------------------
// WMMA GEMM, unit K-stride on both operands:
//   D[m,n] = scale * sum_k A[m + 0*k? no: A[m*aSm + k] * B[n*bSn + k]]
//            (+ bias[m]) (+ residual[m*oSm + n*oSn])
// Block = 256 threads = 8 wave32; wave w owns D rows [*, 16) stripe of a
// 128-row block tile and NSUB=4 adjacent 16-col subtiles (16x64 of D).
// Per 32-K step: 2x b128 A loads (reused 4x), 2x b128 B loads + 1 WMMA per
// subtile.  Fragment layouts used here (contiguous K, same order for A and B):
//   A: lane 0-15 -> M=lane,    halves i: K=kk+i
//      lane16-31 -> M=lane-16, halves i: K=kk+16+i
//   B: lane 0-15 -> N=lane,    halves i: K=kk+i
//      lane16-31 -> N=lane-16, halves i: K=kk+16+i
//   D: element r -> M = r + 8*(lane>=16), N = lane&15
// ---------------------------------------------------------------------------
__global__ void wmma_gemm16(
    const _Float16* __restrict__ A, long long aSm, long long aSb,
    const _Float16* __restrict__ B, long long bSn, long long bSb,
    const float* __restrict__ bias,                      // [M] or null
    const float* __restrict__ residual, long long rSb,   // out layout, or null
    float scale,
    float* __restrict__ outF, _Float16* __restrict__ outH,  // one non-null
    long long oSm, long long oSn, long long oSb,
    int M, int N, int K) {
  const int lane = threadIdx.x & 31;
  const int wave = threadIdx.x >> 5;
  const int m0 = ((int)blockIdx.y * 8 + wave) * 16;
  const int n0 = (int)blockIdx.x * (16 * NSUB);
  const int b  = (int)blockIdx.z;
  if (m0 >= M || n0 >= N) return;   // wave-uniform; EXEC all-ones for WMMA

  A += (long long)b * aSb;
  B += (long long)b * bSb;

  const int aKlo = (lane < 16) ? 0 : 16;
  const int bKlo = (lane < 16) ? 0 : 16;
  const _Float16* __restrict__ aP = A + (long long)(m0 + (lane & 15)) * aSm + aKlo;
  const _Float16* __restrict__ bP = B + (long long)(n0 + (lane & 15)) * bSn + bKlo;

  v8f acc[NSUB];
#pragma unroll
  for (int s = 0; s < NSUB; ++s) acc[s] = (v8f){};

  for (int kk = 0; kk < K; kk += 32) {
    v8h a0 = *(const v8h*)(aP + kk);
    v8h a1 = *(const v8h*)(aP + kk + 8);
    v16h af = __builtin_shufflevector(a0, a1, 0, 1, 2, 3, 4, 5, 6, 7,
                                      8, 9, 10, 11, 12, 13, 14, 15);
    if (kk + 32 < K) {   // gfx1250 global_prefetch_b8 of next K tile
      __builtin_prefetch((const void*)(aP + kk + 32), 0, 1);
      __builtin_prefetch((const void*)(bP + kk + 32), 0, 1);
    }
#pragma unroll
    for (int s = 0; s < NSUB; ++s) {
      const _Float16* bp = bP + (long long)s * 16 * bSn + kk;
      v8h b0 = *(const v8h*)(bp);
      v8h b1 = *(const v8h*)(bp + 8);
      v16h bf = __builtin_shufflevector(b0, b1, 0, 1, 2, 3, 4, 5, 6, 7,
                                        8, 9, 10, 11, 12, 13, 14, 15);
      acc[s] = __builtin_amdgcn_wmma_f32_16x16x32_f16(
          /*neg_a=*/false, af, /*neg_b=*/false, bf,
          /*c_mod=*/(short)0, acc[s], /*reuse_a=*/false, /*reuse_b=*/false);
    }
  }

  const float* res = residual ? residual + (long long)b * rSb : nullptr;
  float*    oF = outF ? outF + (long long)b * oSb : nullptr;
  _Float16* oH = outH ? outH + (long long)b * oSb : nullptr;
  const int mb = m0 + ((lane >= 16) ? 8 : 0);

#pragma unroll
  for (int s = 0; s < NSUB; ++s) {
    const long long n = n0 + s * 16 + (lane & 15);
    if (oH && oSm == 1 && !res) {
      // 8 contiguous M elements per lane -> single 16B store
      v8h pk;
#pragma unroll
      for (int r = 0; r < 8; ++r) {
        float v = acc[s][r] * scale;
        if (bias) v += bias[mb + r];
        pk[r] = (_Float16)v;
      }
      *(v8h*)(oH + mb + n * oSn) = pk;
    } else {
#pragma unroll
      for (int r = 0; r < 8; ++r) {
        const long long m = mb + r;
        float v = acc[s][r] * scale;
        if (bias) v += bias[m];
        const long long idx = m * oSm + n * oSn;
        if (res) v += res[idx];
        if (oF) oF[idx] = v;
        else    oH[idx] = (_Float16)v;
      }
    }
  }
}

// ---------------------------------------------------------------------------
// Host side
// ---------------------------------------------------------------------------
extern "C" void kernel_launch(void* const* d_in, const int* in_sizes, int n_in,
                              void* d_out, int out_size, void* d_ws, size_t ws_size,
                              hipStream_t stream) {
  (void)in_sizes; (void)n_in; (void)out_size; (void)ws_size;
  const float* x  = (const float*)d_in[0];
  const float* gw = (const float*)d_in[1];
  const float* gb = (const float*)d_in[2];
  const float* wq = (const float*)d_in[3];
  const float* bq = (const float*)d_in[4];
  const float* wk = (const float*)d_in[5];
  const float* bk = (const float*)d_in[6];
  const float* wv = (const float*)d_in[7];
  const float* bv = (const float*)d_in[8];
  const float* wp = (const float*)d_in[9];
  const float* bp = (const float*)d_in[10];
  float* out = (float*)d_out;

  // ---- workspace carve-up -------------------------------------------------
  char* base = (char*)d_ws;
  size_t off = 0;
  auto take = [&](size_t bytes) -> char* {
    char* p = base + off;
    off = (off + bytes + 255) & ~(size_t)255;
    return p;
  };
  const long long actEls = (long long)BATCH * CCH * HW;      // 4M
  const long long attEls = (long long)BATCH * HW * HW;       // 32M

  float*    mean = (float*)take(16 * sizeof(float));
  float*    rstd = (float*)take(16 * sizeof(float));
  _Float16* wqh  = (_Float16*)take((size_t)CCH * CCH * 2);
  _Float16* wkh  = (_Float16*)take((size_t)CCH * CCH * 2);
  _Float16* wvh  = (_Float16*)take((size_t)CCH * CCH * 2);
  _Float16* wph  = (_Float16*)take((size_t)CCH * CCH * 2);
  _Float16* xnT  = (_Float16*)take((size_t)actEls * 2);  // [B,HW,C]
  _Float16* qT   = (_Float16*)take((size_t)actEls * 2);  // [B,HW,C]
  _Float16* kT   = (_Float16*)take((size_t)actEls * 2);  // [B,HW,C]
  _Float16* vA   = (_Float16*)take((size_t)actEls * 2);  // [B,C,HW]
  _Float16* oT   = (_Float16*)take((size_t)actEls * 2);  // [B,HW,C]
  float*    Sb   = (float*)take((size_t)attEls * 4);     // [B,i,j] logits
  _Float16* Pb   = (_Float16*)take((size_t)attEls * 2);  // [B,i,j] probs

  const long long actB = (long long)CCH * HW;   // batch stride, activations
  const long long attB = (long long)HW * HW;    // batch stride, attention
  const float scale = 0.044194173824159216f;    // 512^-0.5

  // ---- 1) weight casts ----------------------------------------------------
  {
    int n = CCH * CCH, blk = 256, grd = (n + blk - 1) / blk;
    cast_f32_f16<<<grd, blk, 0, stream>>>(wq, wqh, n);
    cast_f32_f16<<<grd, blk, 0, stream>>>(wk, wkh, n);
    cast_f32_f16<<<grd, blk, 0, stream>>>(wv, wvh, n);
    cast_f32_f16<<<grd, blk, 0, stream>>>(wp, wph, n);
  }

  // ---- 2) GroupNorm (emit token-major f16) --------------------------------
  gn_stats<<<BATCH * NGRP, 256, 0, stream>>>(x, mean, rstd);
  gn_apply_t<<<(unsigned)(actEls / 256), 256, 0, stream>>>(x, mean, rstd, gw, gb, xnT);

  // ---- 3) q / k / v GEMMs: [512x512] x [512x4096] + bias ------------------
  // A = W (row-major, aSm=512); B[k=c,n=t] = xnT[t*512+c] (bSn=512).
  {
    dim3 g(HW / (16 * NSUB), CCH / 128, BATCH);   // (64, 4, 2)
    // q -> qT [t, o] : idx = n*512 + m  (oSm=1, oSn=512), vector f16 stores
    wmma_gemm16<<<g, 256, 0, stream>>>(wqh, CCH, 0,  xnT, CCH, actB,
                                       bq, nullptr, 0, 1.0f,
                                       nullptr, qT, 1, CCH, actB, CCH, HW, CCH);
    wmma_gemm16<<<g, 256, 0, stream>>>(wkh, CCH, 0,  xnT, CCH, actB,
                                       bk, nullptr, 0, 1.0f,
                                       nullptr, kT, 1, CCH, actB, CCH, HW, CCH);
    // v -> vA [o, t] : idx = m*4096 + n (coalesced across lanes)
    wmma_gemm16<<<g, 256, 0, stream>>>(wvh, CCH, 0,  xnT, CCH, actB,
                                       bv, nullptr, 0, 1.0f,
                                       nullptr, vA, HW, 1, actB, CCH, HW, CCH);
  }

  // ---- 4) S = scale * Q^T K : A[m=i,k=c]=qT[i*512+c], B[k=c,n=j]=kT[j*512+c]
  {
    dim3 g(HW / (16 * NSUB), HW / 128, BATCH);    // (64, 32, 2)
    wmma_gemm16<<<g, 256, 0, stream>>>(qT, CCH, actB,  kT, CCH, actB,
                                       nullptr, nullptr, 0, scale,
                                       Sb, nullptr, HW, 1, attB, HW, HW, CCH);
  }

  // ---- 5) softmax over j (LDS-staged via async loads) ---------------------
  softmax_rows<<<BATCH * HW, 256, 0, stream>>>(Sb, Pb);

  // ---- 6) O = V P^T : A[m=c,k=j]=vA[c*4096+j], B[k=j,n=i]=Pb[i*4096+j] ----
  {
    dim3 g(HW / (16 * NSUB), CCH / 128, BATCH);
    // out -> oT [i, c] : idx = n*512 + m (oSm=1, oSn=512), vector f16 stores
    wmma_gemm16<<<g, 256, 0, stream>>>(vA, HW, actB,  Pb, HW, attB,
                                       nullptr, nullptr, 0, 1.0f,
                                       nullptr, oT, 1, CCH, actB, CCH, HW, HW);
  }

  // ---- 7) proj + bias + residual, f32 out [B,C,HW] ------------------------
  {
    dim3 g(HW / (16 * NSUB), CCH / 128, BATCH);
    wmma_gemm16<<<g, 256, 0, stream>>>(wph, CCH, 0,  oT, CCH, actB,
                                       bp, x, actB, 1.0f,
                                       out, nullptr, HW, 1, actB, CCH, HW, CCH);
  }
}
